// TreeLSTMSentiment_82222853915260
// MI455X (gfx1250) — compile-verified, matrix operand order checked by reading, not run on hardware
//
#include <hip/hip_runtime.h>

// ---------------------------------------------------------------------------
// TreeLSTM sentiment for MI455X (gfx1250, wave32, WMMA).
// B=256 trees, N=128 nodes, D=H=300, L=5.
//
//  K1: shuffle U_iou/U_f AND [W_iou|W_f] into bf16 fragment-major layout
//      [nt][kt][lane][e]  (one lane's 16 elements contiguous -> b128 loads)
//  K2: proj GEMM  x_iou/x_f = emb @ [W_iou|W_f] + b : A staged in LDS as
//      f32 quads (b128 load -> convertvector -> 2x v_cvt_pk_bf16_f32 ->
//      ds_store_b64), B fragments streamed from L2 as 32B/lane b128 pairs.
//  K3: persistent recurrence: 16 WGs x 16 trees, 128 serial steps,
//      two fully K-unrolled WMMA GEMMs per step; all fills/activations
//      quad-vectorized with ext-vector types (everything 16B aligned).
//  K4: logits + log_softmax
//
// 16-bit fragment layout (ISA 7.12.2): element e of lane (lhi*16+mn) holds
// k = 16*(e>=8) + 8*lhi + (e&7); any k%4==0 quad is 2 adjacent dwords of
// one lane's block, 8B aligned.
// ---------------------------------------------------------------------------

typedef __attribute__((ext_vector_type(16))) __bf16 v16bf;
typedef __attribute__((ext_vector_type(4)))  __bf16 v4bf;
typedef __attribute__((ext_vector_type(8)))  float  v8f;
typedef __attribute__((ext_vector_type(4)))  float  v4f;
typedef __attribute__((ext_vector_type(2)))  unsigned int v2u;

#define BATCH 256
#define NNODE 128
#define DIN   300
#define HID   300
#define NT_IOU 57          // 912 cols (U_iou)
#define NT_F   19          // 304 cols (U_f)
#define NT_W   80          // 1280 cols ([W_iou|W_f] padded for 10 N-blocks)
#define FRAG_U32 8         // uints per lane per fragment (16 bf16)

// uint index (2 bf16/uint) of local k (0..31) for row/col `mn` in tile `tile`
__device__ __forceinline__ int frag_uidx(int tile, int klocal, int mn) {
    int lhi  = (klocal >> 3) & 1;
    int e    = (klocal & 7) + ((klocal & 16) ? 8 : 0);
    int lane = lhi * 16 + mn;
    return (tile * 32 + lane) * FRAG_U32 + (e >> 1);
}

// 4 consecutive-k f32 -> bf16 (k%4==0): 2x v_cvt_pk_bf16_f32 + ds_store_b64
__device__ __forceinline__ void store_quad(unsigned int* dst, v4f f) {
    union { v4bf h; v2u u; } p;
    p.h = __builtin_convertvector(f, v4bf);
    *reinterpret_cast<v2u*>(dst) = p.u;
}

__device__ __forceinline__ float dev_sigmoid(float x) {
    return 1.0f / (1.0f + __expf(-x));
}

// ---------------------------------------------------------------------------
// K1: shuffle all weights fp32 -> bf16 fragment-major  [nt][kt][lane][e]
// ---------------------------------------------------------------------------
__global__ void __launch_bounds__(256) convert_weights_kernel(
    const float* __restrict__ U_iou, const float* __restrict__ U_f,
    const float* __restrict__ W_iou, const float* __restrict__ W_f,
    unsigned short* __restrict__ UbIou_u, unsigned short* __restrict__ UbF_u,
    unsigned short* __restrict__ Wb_u) {
    __bf16* UbIou = reinterpret_cast<__bf16*>(UbIou_u);
    __bf16* UbF   = reinterpret_cast<__bf16*>(UbF_u);
    __bf16* Wb    = reinterpret_cast<__bf16*>(Wb_u);
    long idx = (long)blockIdx.x * 256 + threadIdx.x;

    int rem  = (int)(idx % (10 * 512));
    int nt   = (int)(idx / (10 * 512));
    int kt   = rem / 512, le = rem % 512;
    int lane = le >> 4, e = le & 15;
    int lhi  = lane >> 4, lno = lane & 15;
    int k    = kt * 32 + ((e >= 8) ? 16 : 0) + lhi * 8 + (e & 7);
    int col  = nt * 16 + lno;

    if (idx < (long)NT_IOU * 10 * 512) {
        float v = (k < HID && col < 3 * HID) ? U_iou[k * (3 * HID) + col] : 0.0f;
        UbIou[idx] = (__bf16)v;
    }
    if (idx < (long)NT_F * 10 * 512) {
        float v = (k < HID && col < HID) ? U_f[k * HID + col] : 0.0f;
        UbF[idx] = (__bf16)v;
    }
    if (idx < (long)NT_W * 10 * 512) {
        float v = 0.0f;
        if (k < DIN) {
            if (col < 3 * HID)       v = W_iou[k * (3 * HID) + col];
            else if (col < 4 * HID)  v = W_f[k * HID + (col - 3 * HID)];
        }
        Wb[idx] = (__bf16)v;
    }
}

// ---------------------------------------------------------------------------
// K2: projection GEMM.  [32768 x 300] @ [300 x 1200] -> x_iou | x_f
// block: 256 thr (8 waves), tile 64(M) x 128(N), wave tile 32x32 (2x2 WMMA)
// ---------------------------------------------------------------------------
__global__ void __launch_bounds__(256) proj_gemm_kernel(
    const float* __restrict__ emb,
    const unsigned short* __restrict__ Wb_u,
    const float* __restrict__ b_iou, const float* __restrict__ b_f,
    float* __restrict__ x_iou, float* __restrict__ x_f) {
    __shared__ __attribute__((aligned(32))) unsigned int sA_u[4 * 32 * FRAG_U32]; // 4 KB
    const v16bf* aFrag = reinterpret_cast<const v16bf*>(sA_u);
    const v16bf* wbF   = reinterpret_cast<const v16bf*>(Wb_u);

    const int tid  = threadIdx.x;
    const int wave = tid >> 5, lane = tid & 31;
    const int lhi  = lane >> 4, lno = lane & 15;
    const int mt0  = (wave >> 2) * 2;        // m-tile base: 0 / 2
    const int nt0  = (wave & 3) * 2;         // n-tile base: 0 / 2 / 4 / 6
    const long rowBase = (long)blockIdx.x * 64;
    const int  colBase = blockIdx.y * 128;
    const int  ntBase  = blockIdx.y * 8;     // global n-tile base

    v8f acc[2][2] = {};

    for (int kt = 0; kt < 10; ++kt) {
        const int k0 = kt * 32;
        // ---- stage A [64 x 32] as f32 quads -> fragment-major bf16
        #pragma unroll
        for (int i = 0; i < 2; ++i) {
            int p = tid + i * 256;           // 512 quads
            int r = p >> 3, k = (p & 7) * 4;
            int gk = k0 + k;
            v4f f = {0.f, 0.f, 0.f, 0.f};
            if (gk < DIN)    // DIN % 4 == 0: quads are fully in or fully out
                f = *reinterpret_cast<const v4f*>(&emb[(rowBase + r) * DIN + gk]);
            store_quad(&sA_u[frag_uidx(r >> 4, k, r & 15)], f);
        }
        __syncthreads();

        v16bf af[2], bfr[2];
        #pragma unroll
        for (int t = 0; t < 2; ++t) {
            af[t]  = aFrag[(mt0 + t) * 32 + lane];
            bfr[t] = wbF[((ntBase + nt0 + t) * 10 + kt) * 32 + lane];
        }
        #pragma unroll
        for (int mi = 0; mi < 2; ++mi)
            #pragma unroll
            for (int ni = 0; ni < 2; ++ni)
                acc[mi][ni] = __builtin_amdgcn_wmma_f32_16x16x32_bf16(
                    false, af[mi], false, bfr[ni], (short)0, acc[mi][ni], false, false);
        __syncthreads();
    }

    // ---- epilogue: D layout  m = lhi*8 + r, n = lno
    #pragma unroll
    for (int mi = 0; mi < 2; ++mi)
        #pragma unroll
        for (int ni = 0; ni < 2; ++ni)
            #pragma unroll
            for (int r = 0; r < 8; ++r) {
                long row = rowBase + (mt0 + mi) * 16 + lhi * 8 + r;
                int  col = colBase + (nt0 + ni) * 16 + lno;
                float v = acc[mi][ni][r];
                if (col < 3 * HID)       x_iou[row * (3 * HID) + col] = v + b_iou[col];
                else if (col < 4 * HID)  x_f[row * HID + (col - 3 * HID)] = v + b_f[col - 3 * HID];
            }
}

// ---------------------------------------------------------------------------
// K3: persistent recurrence.  grid=16 WGs, 512 thr (16 waves), 16 trees/WG.
// ---------------------------------------------------------------------------
__global__ void __launch_bounds__(512) tree_scan_kernel(
    const float* __restrict__ x_iou,            // [B*N, 900]
    const float* __restrict__ x_f,              // [B*N, 300]
    const int*   __restrict__ parents,          // [B, N]
    const unsigned short* __restrict__ UbIou_u, // bf16 frag-major [57][10][512]
    const unsigned short* __restrict__ UbF_u,   // bf16 frag-major [19][10][512]
    float* __restrict__ h_sum,                  // [B*N, 300]
    float* __restrict__ c_sum,                  // [B*N, 300]
    float* __restrict__ rootH) {                // [B, 300]
    const v16bf* biF = reinterpret_cast<const v16bf*>(UbIou_u);
    const v16bf* bfF = reinterpret_cast<const v16bf*>(UbF_u);

    const int b0   = blockIdx.x * 16;
    const int tid  = threadIdx.x;
    const int wave = tid >> 5, lane = tid & 31;
    const int lhi  = lane >> 4, lno = lane & 15;

    __shared__ __attribute__((aligned(32))) unsigned int sA_u[10 * 32 * FRAG_U32]; // 10 KB
    __shared__ __attribute__((aligned(32))) unsigned int sH_u[10 * 32 * FRAG_U32]; // 10 KB
    __shared__ __attribute__((aligned(16))) float sIOU[16][NT_IOU * 16];           // 58.4 KB
    __shared__ int   sPar[16];
    const v16bf* aF = reinterpret_cast<const v16bf*>(sA_u);
    const v16bf* hF = reinterpret_cast<const v16bf*>(sH_u);

    // ---- zero this block's recurrent state slab (b128 stores)
    const v4f zero4 = {0.f, 0.f, 0.f, 0.f};
    for (int e = tid; e < 16 * NNODE * (HID / 4); e += 512) {
        long o = (long)(b0 * NNODE) * HID + (long)e * 4;  // contiguous slab
        *reinterpret_cast<v4f*>(&h_sum[o]) = zero4;
        *reinterpret_cast<v4f*>(&c_sum[o]) = zero4;
    }
    // zero K padding (cols 300..319 -> kt=9, klocal 12..28 quads)
    for (int e = tid; e < 16 * 5; e += 512) {
        int m = e / 5, klocal = 12 + (e % 5) * 4;
        v2u z = {0u, 0u};
        *reinterpret_cast<v2u*>(&sA_u[frag_uidx(9, klocal, m)]) = z;
        *reinterpret_cast<v2u*>(&sH_u[frag_uidx(9, klocal, m)]) = z;
    }
    __threadfence();
    __syncthreads();

    for (int j = NNODE - 1; j >= 0; --j) {
        if (tid < 16) sPar[tid] = parents[(b0 + tid) * NNODE + j];
        // ---- load h_sum rows for node j (16 x 300) as f32 quads, frag-major
        for (int e = tid; e < 16 * 75; e += 512) {
            int m = e / 75, col = (e % 75) * 4;
            v4f h4 = *reinterpret_cast<const v4f*>(
                &h_sum[((long)(b0 + m) * NNODE + j) * HID + col]);
            store_quad(&sA_u[frag_uidx(col >> 5, col & 31, m)], h4);
        }
        // prefetch next step's x_iou rows (global_prefetch_b8)
        if (j > 0 && tid < 16 * 8) {
            int m = tid >> 3, ch = tid & 7;
            __builtin_prefetch(
                &x_iou[((long)(b0 + m) * NNODE + (j - 1)) * (3 * HID) + ch * 128], 0, 1);
        }
        __syncthreads();

        // ---- GEMM1: sIOU = hsum(16x320) @ UbIou(320x912), K fully unrolled
        for (int nt = wave; nt < NT_IOU; nt += 16) {
            v8f acc = {};
            #pragma unroll
            for (int kt = 0; kt < 10; ++kt)
                acc = __builtin_amdgcn_wmma_f32_16x16x32_bf16(
                    false, aF[kt * 32 + lane], false, biF[(nt * 10 + kt) * 32 + lane],
                    (short)0, acc, false, false);
            #pragma unroll
            for (int r = 0; r < 8; ++r)
                sIOU[lhi * 8 + r][nt * 16 + lno] = acc[r];
        }
        __syncthreads();

        // ---- activations over col quads; c kept in the u-slot of sIOU
        for (int e = tid; e < 16 * 75; e += 512) {
            int m = e / 75, col = (e % 75) * 4;
            long row = (long)(b0 + m) * NNODE + j;
            const float* xr = &x_iou[row * (3 * HID)];
            v4f iv = *reinterpret_cast<const v4f*>(&sIOU[m][col]) +
                     *reinterpret_cast<const v4f*>(&xr[col]);
            v4f ov = *reinterpret_cast<const v4f*>(&sIOU[m][col + HID]) +
                     *reinterpret_cast<const v4f*>(&xr[col + HID]);
            v4f uv = *reinterpret_cast<const v4f*>(&sIOU[m][col + 2 * HID]) +
                     *reinterpret_cast<const v4f*>(&xr[col + 2 * HID]);
            v4f cs = *reinterpret_cast<const v4f*>(&c_sum[row * HID + col]);
            v4f cv, hv;
            #pragma unroll
            for (int q = 0; q < 4; ++q) {
                float ig = dev_sigmoid(iv[q]);
                float og = dev_sigmoid(ov[q]);
                float ug = tanhf(uv[q]);
                cv[q] = ig * ug + cs[q];
                hv[q] = og * tanhf(cv[q]);
            }
            *reinterpret_cast<v4f*>(&sIOU[m][col + 2 * HID]) = cv;  // c for gemm2
            store_quad(&sH_u[frag_uidx(col >> 5, col & 31, m)], hv);
            if (j == 0) {
                *reinterpret_cast<v4f*>(&rootH[(long)(b0 + m) * HID + col]) = hv;
            } else {
                float* hp = &h_sum[((long)(b0 + m) * NNODE + sPar[m]) * HID + col];
                v4f old = *reinterpret_cast<const v4f*>(hp);
                *reinterpret_cast<v4f*>(hp) = old + hv;
            }
        }
        __threadfence();
        __syncthreads();

        // ---- GEMM2: fIn = h(16x320) @ UbF(320x304); f-gate + c_sum scatter
        if (j > 0) {
            for (int nt = wave; nt < NT_F; nt += 16) {
                v8f acc = {};
                #pragma unroll
                for (int kt = 0; kt < 10; ++kt)
                    acc = __builtin_amdgcn_wmma_f32_16x16x32_bf16(
                        false, hF[kt * 32 + lane], false, bfF[(nt * 10 + kt) * 32 + lane],
                        (short)0, acc, false, false);
                #pragma unroll
                for (int r = 0; r < 8; ++r) {
                    int m = lhi * 8 + r, col = nt * 16 + lno;
                    if (col < HID) {
                        long prow = (long)(b0 + m) * NNODE + sPar[m];
                        float f = dev_sigmoid(x_f[prow * HID + col] + acc[r]);
                        c_sum[prow * HID + col] += f * sIOU[m][col + 2 * HID];
                    }
                }
            }
            __threadfence();
        }
        __syncthreads();
    }
}

// ---------------------------------------------------------------------------
// K4: logits + log_softmax
// ---------------------------------------------------------------------------
__global__ void __launch_bounds__(256) classify_kernel(
    const float* __restrict__ rootH, const float* __restrict__ W_out,
    const float* __restrict__ b_out, float* __restrict__ out) {
    int b = blockIdx.x * blockDim.x + threadIdx.x;
    if (b >= BATCH) return;
    float logit[5];
    #pragma unroll
    for (int l = 0; l < 5; ++l) logit[l] = b_out[l];
    for (int k = 0; k < HID; ++k) {
        float h = rootH[(long)b * HID + k];
        #pragma unroll
        for (int l = 0; l < 5; ++l) logit[l] += h * W_out[k * 5 + l];
    }
    float mx = logit[0];
    #pragma unroll
    for (int l = 1; l < 5; ++l) mx = fmaxf(mx, logit[l]);
    float s = 0.0f;
    #pragma unroll
    for (int l = 0; l < 5; ++l) s += __expf(logit[l] - mx);
    float lse = __logf(s) + mx;
    #pragma unroll
    for (int l = 0; l < 5; ++l) out[b * 5 + l] = logit[l] - lse;
}

// ---------------------------------------------------------------------------
extern "C" void kernel_launch(void* const* d_in, const int* in_sizes, int n_in,
                              void* d_out, int out_size, void* d_ws, size_t ws_size,
                              hipStream_t stream) {
    const float* emb   = (const float*)d_in[0];
    const int*   par   = (const int*)d_in[1];
    const float* W_iou = (const float*)d_in[2];
    const float* b_iou = (const float*)d_in[3];
    const float* U_iou = (const float*)d_in[4];
    const float* W_f   = (const float*)d_in[5];
    const float* b_f   = (const float*)d_in[6];
    const float* U_f   = (const float*)d_in[7];
    const float* W_out = (const float*)d_in[8];
    const float* b_out = (const float*)d_in[9];
    float* out = (float*)d_out;
    (void)in_sizes; (void)n_in; (void)out_size; (void)ws_size;

    size_t off = 0;
    auto take = [&](size_t bytes) -> char* {
        char* p = (char*)d_ws + off;
        off += (bytes + 255) & ~(size_t)255;
        return p;
    };
    const size_t ROWS = (size_t)BATCH * NNODE;          // 32768
    float*          x_iou = (float*)take(ROWS * 3 * HID * 4);
    float*          x_f   = (float*)take(ROWS * HID * 4);
    float*          h_sum = (float*)take(ROWS * HID * 4);
    float*          c_sum = (float*)take(ROWS * HID * 4);
    float*          rootH = (float*)take((size_t)BATCH * HID * 4);
    unsigned short* UbIou = (unsigned short*)take((size_t)NT_IOU * 10 * 512 * 2);
    unsigned short* UbF   = (unsigned short*)take((size_t)NT_F * 10 * 512 * 2);
    unsigned short* Wb    = (unsigned short*)take((size_t)NT_W * 10 * 512 * 2);

    // K1: weight shuffle/convert (covers the largest buffer: NT_W tiles)
    {
        int total = NT_W * 10 * 512;
        convert_weights_kernel<<<(total + 255) / 256, 256, 0, stream>>>(
            U_iou, U_f, W_iou, W_f, UbIou, UbF, Wb);
    }
    // K2: projection GEMM: M tiles = 32768/64 = 512, N tiles = 1280/128 = 10
    proj_gemm_kernel<<<dim3(512, 10), 256, 0, stream>>>(
        emb, Wb, b_iou, b_f, x_iou, x_f);
    // K3: persistent recurrence, 16 trees per WG
    tree_scan_kernel<<<BATCH / 16, 512, 0, stream>>>(
        x_iou, x_f, par, UbIou, UbF, h_sum, c_sum, rootH);
    // K4: classifier
    classify_kernel<<<1, 256, 0, stream>>>(rootH, W_out, b_out, out);
}
